// MultiHeadAttention_48395691491531
// MI455X (gfx1250) — matrix operand order repaired and checked by reading, not compiled, not run on hardware
//
#include <hip/hip_runtime.h>
#include <hip/hip_bf16.h>
#include <math.h>

// ---------------------------------------------------------------------------
// MHA + RoPE for MI455X (gfx1250): all GEMMs via v_wmma_f32_16x16x32_bf16.
// GEMM staging: A tile via async global->LDS DMA (ASYNCcnt), B tile via the
// Tensor Data Mover (TENSORcnt), double-buffered. S=2048, E=2048, H=16, D=128.
// ---------------------------------------------------------------------------

#define S_LEN 2048
#define EMB   2048
#define NH    16
#define HD    128

typedef __attribute__((ext_vector_type(16))) __bf16 v16bf;
typedef __attribute__((ext_vector_type(8)))  __bf16 v8bf;
typedef __attribute__((ext_vector_type(4)))  __bf16 v4bf;
typedef __attribute__((ext_vector_type(8)))  float  v8f;
typedef __attribute__((ext_vector_type(4)))  float  v4f;

typedef __attribute__((address_space(3))) __bf16 lds_bf16_t;

#if defined(__has_builtin)
#if __has_builtin(__builtin_amdgcn_tensor_load_to_lds)
#define USE_TDM 1
#endif
#endif
#ifndef USE_TDM
#define USE_TDM 0
#endif

// A-matrix 16x32 bf16 fragment (ISA 7.12.2): lane ll (row) holds
// K = {8*hi .. 8*hi+7} in v0..v3 and K = {16+8*hi ..} in v4..v7.
static __device__ __forceinline__ v16bf load_afrag(const __bf16* row, int hi) {
    v8bf lo = *(const v8bf*)(row + 8 * hi);
    v8bf up = *(const v8bf*)(row + 16 + 8 * hi);
    return __builtin_shufflevector(lo, up, 0,1,2,3,4,5,6,7,8,9,10,11,12,13,14,15);
}
// B-matrix 32x16 bf16 fragment: lane ll (column) holds K = {8*hi .. 8*hi+7}
// in v0..v3 and K = {16+8*hi ..} in v4..v7 (same order as A).
static __device__ __forceinline__ v16bf load_bfrag(const __bf16* row, int hi) {
    v8bf lo = *(const v8bf*)(row + 8 * hi);
    v8bf up = *(const v8bf*)(row + 16 + 8 * hi);
    return __builtin_shufflevector(lo, up, 0,1,2,3,4,5,6,7,8,9,10,11,12,13,14,15);
}

static __device__ __forceinline__ v8f wmma_bf16(v16bf a, v16bf b, v8f c) {
    return __builtin_amdgcn_wmma_f32_16x16x32_bf16(
        /*neg_a=*/false, a, /*neg_b=*/false, b,
        /*c_mod=*/(short)0, c, /*reuse_a=*/false, /*reuse_b=*/false);
}

// Async DMA: 16B per lane, global -> LDS, tracked by ASYNCcnt.
static __device__ __forceinline__ void async_load_b128(unsigned lds_addr,
                                                       unsigned voff,
                                                       const __bf16* sbase) {
    asm volatile("global_load_async_to_lds_b128 %0, %1, %2"
                 :: "v"(lds_addr), "v"(voff), "s"(sbase) : "memory");
}

#if USE_TDM
typedef __attribute__((ext_vector_type(4))) unsigned tdm_v4u;
typedef __attribute__((ext_vector_type(8))) int      tdm_v8i;
typedef __attribute__((ext_vector_type(4))) int      tdm_v4i;

// TDM: load one 128-row x 32-elem bf16 tile (row-major, row stride Kdim elems)
// into LDS with 16B padding after each 64B row chunk -> 80B LDS row stride.
// D# per ISA 08_async_tensor §8: group0 {count=1, lds_addr, global_addr,
// type=2}; group1 {data_size=2B, pad_enable, pad_interval=16dw, pad_amount=4dw,
// tensor_dim0=Kdim, tensor_dim1=rows, tile_dim0=32, tile_dim1=128,
// tensor_dim0_stride=Kdim}. Remaining groups zero (2D tensor).
static __device__ __forceinline__ void tdm_load_tile_128x32(
    const __bf16* tile_start, unsigned lds_byte_addr, unsigned Kdim,
    unsigned rows) {
    unsigned long long ga = (unsigned long long)tile_start;
    tdm_v4u g0;
    g0.x = 1u;                                    // count=1, user descriptor
    g0.y = lds_byte_addr;                         // bits [63:32]
    g0.z = (unsigned)ga;                          // bits [95:64]
    g0.w = (unsigned)(ga >> 32) | (2u << 30);     // bits [120:96] + type=2
    tdm_v8i g1;
    g1[0] = (int)((1u << 16) |                    // data_size = 2 bytes
                  (1u << 20) |                    // pad_enable
                  (3u << 22) |                    // pad_interval: 16 DWORDs
                  (3u << 25));                    // pad_amount:   4 DWORDs
    g1[1] = (int)(Kdim << 16);                    // tensor_dim0[15:0]
    g1[2] = (int)((Kdim >> 16) | ((rows & 0xffffu) << 16)); // dim0 hi, dim1 lo
    g1[3] = (int)((rows >> 16) | (32u << 16));    // dim1 hi, tile_dim0=32
    g1[4] = 128;                                  // tile_dim1=128, tile_dim2=0
    g1[5] = (int)Kdim;                            // tensor_dim0_stride lo32
    g1[6] = 0;                                    // stride hi, dim1_stride lo
    g1[7] = 0;
    tdm_v4i z4 = {0, 0, 0, 0};
    tdm_v8i z8 = {0, 0, 0, 0, 0, 0, 0, 0};
    // 6-arg form (clang-23 / therock-10.0): (g0, g1, g2, g3, g4, cpol)
    __builtin_amdgcn_tensor_load_to_lds(g0, g1, z4, z4, z8, 0);
}

#define WAIT_STAGE_PREFETCHED() do {                                  \
        asm volatile("s_wait_asynccnt 0x2" ::: "memory");             \
        __builtin_amdgcn_s_wait_tensorcnt((short)1);                  \
    } while (0)
#define WAIT_STAGE_LAST() do {                                        \
        asm volatile("s_wait_asynccnt 0x0" ::: "memory");             \
        __builtin_amdgcn_s_wait_tensorcnt((short)0);                  \
    } while (0)
#else
#define WAIT_STAGE_PREFETCHED() \
        asm volatile("s_wait_asynccnt 0x4" ::: "memory")
#define WAIT_STAGE_LAST() \
        asm volatile("s_wait_asynccnt 0x0" ::: "memory")
#endif

// ---------------------------------------------------------------------------
// f32 -> bf16 convert (4 elements / thread; n is a multiple of 1024)
// ---------------------------------------------------------------------------
__global__ void cvt_bf16_kernel(const float* __restrict__ in,
                                __bf16* __restrict__ out, int n) {
    int i = (blockIdx.x * blockDim.x + threadIdx.x) * 4;
    if (i < n) {
        v4f v = *(const v4f*)(in + i);
        v4bf o;
        o.x = (__bf16)v.x; o.y = (__bf16)v.y; o.z = (__bf16)v.z; o.w = (__bf16)v.w;
        *(v4bf*)(out + i) = o;
    }
}

// ---------------------------------------------------------------------------
// C[M][N] = A[M][K] * W[N][K]^T + bias[N]   (A,W bf16 row-major; C f32)
// Block tile 128x128, K-step 32, 8 waves (wave grid 2x4, 4x2 WMMA tiles each).
// A tile: 2 async-DMA loads per thread per stage (ASYNCcnt, in-order).
// B tile: one TDM descriptor issued by wave 0 per stage (TENSORcnt, in-order).
// Ping-pong double buffer; last iteration peeled to keep the hot loop
// branch-free.
// ---------------------------------------------------------------------------
__global__ __launch_bounds__(256) void gemm_bf16_wmma(
    const __bf16* __restrict__ A, const __bf16* __restrict__ W,
    const float* __restrict__ bias, float* __restrict__ C,
    int M, int N, int K)
{
    // [buf][ A:128x40 | B:128x40 ]  (rows padded 32->40 elems = 80B stride)
    __shared__ __align__(16) __bf16 smem[2 * 2 * 128 * 40];

    const int tid  = threadIdx.x;
    const int lane = tid & 31;
    const int wave = tid >> 5;
    const int ll   = lane & 15;
    const int hi   = lane >> 4;
    const int wm   = wave >> 2;   // 0..1
    const int wn   = wave & 3;    // 0..3
    const int m0   = blockIdx.y * 128;
    const int n0   = blockIdx.x * 128;

    const unsigned lds_base = (unsigned)(size_t)(lds_bf16_t*)smem;

    // A staging map: 512 16B-slots; thread covers slot tid and tid+256
    // (same seg, row+64).
    const int r0 = tid >> 2;
    const int s0 = tid & 3;
    const unsigned lA0 = lds_base + (unsigned)(r0 * 80 + s0 * 16);
    const unsigned lA1 = lA0 + 64u * 80u;
    const unsigned voffA0 = ((unsigned)(m0 + r0) * (unsigned)K + (unsigned)(s0 * 8)) * 2u;
    const unsigned voffA1 = voffA0 + 64u * (unsigned)K * 2u;
#if !USE_TDM
    const unsigned lB0 = lA0 + 10240u;
    const unsigned lB1 = lA1 + 10240u;
    const unsigned voffB0 = ((unsigned)(n0 + r0) * (unsigned)K + (unsigned)(s0 * 8)) * 2u;
    const unsigned voffB1 = voffB0 + 64u * (unsigned)K * 2u;
#endif

    auto issue_stage = [&](int k0, int buf) {
        const __bf16* aK = A + k0;
        const unsigned bo = (unsigned)buf * 20480u;   // bytes per buffer
        async_load_b128(lA0 + bo, voffA0, aK);
        async_load_b128(lA1 + bo, voffA1, aK);
#if USE_TDM
        if (wave == 0)
            tdm_load_tile_128x32(W + (unsigned)n0 * (unsigned)K + k0,
                                 lds_base + bo + 10240u,
                                 (unsigned)K, (unsigned)N);
#else
        const __bf16* wK = W + k0;
        async_load_b128(lB0 + bo, voffB0, wK);
        async_load_b128(lB1 + bo, voffB1, wK);
#endif
    };

    v8f acc[4][2];
#pragma unroll
    for (int mt = 0; mt < 4; ++mt)
#pragma unroll
        for (int nt = 0; nt < 2; ++nt) acc[mt][nt] = (v8f)0.0f;

    auto compute_stage = [&](int buf) {
        const __bf16* sa = smem + (unsigned)buf * 10240u;   // elements
        const __bf16* sb = sa + 5120;
        v16bf af[4], bfr[2];
#pragma unroll
        for (int mt = 0; mt < 4; ++mt)
            af[mt] = load_afrag(&sa[(wm * 64 + mt * 16 + ll) * 40], hi);
#pragma unroll
        for (int nt = 0; nt < 2; ++nt)
            bfr[nt] = load_bfrag(&sb[(wn * 32 + nt * 16 + ll) * 40], hi);
#pragma unroll
        for (int mt = 0; mt < 4; ++mt)
#pragma unroll
            for (int nt = 0; nt < 2; ++nt)
                acc[mt][nt] = wmma_bf16(af[mt], bfr[nt], acc[mt][nt]);
    };

    const int nsteps = K >> 5;
    issue_stage(0, 0);

    int i = 0;
    for (; i < nsteps - 1; ++i) {
        // Target buffer was last read two stages ago; its end-of-iteration
        // barrier has been passed, so the overwrite is safe.
        issue_stage((i + 1) << 5, (i + 1) & 1);
        WAIT_STAGE_PREFETCHED();   // my stage-i DMA landed (in-order)
        __syncthreads();           // everyone's stage-i DMA landed
        compute_stage(i & 1);
        __syncthreads();           // buffer free for reuse
    }
    WAIT_STAGE_LAST();
    __syncthreads();
    compute_stage(i & 1);

    // C/D layout: VGPR r -> row (r + 8*hi) in tile, col = ll. 32-bit indexing.
#pragma unroll
    for (int mt = 0; mt < 4; ++mt)
#pragma unroll
        for (int nt = 0; nt < 2; ++nt)
#pragma unroll
            for (int r = 0; r < 8; ++r) {
                unsigned row = (unsigned)(m0 + wm * 64 + mt * 16 + r + 8 * hi);
                unsigned col = (unsigned)(n0 + wn * 32 + nt * 16 + ll);
                C[row * (unsigned)N + col] = acc[mt][nt][r] + bias[col];
            }
}

// ---------------------------------------------------------------------------
// RoPE + scale + pack: f32 [S][E] -> bf16 [H][S][D]
// ---------------------------------------------------------------------------
__global__ void rope_pack_kernel(const float* __restrict__ X,
                                 __bf16* __restrict__ out, float scale) {
    int idx = blockIdx.x * blockDim.x + threadIdx.x;   // H*S*64
    int d = idx & 63;
    int s = (idx >> 6) & (S_LEN - 1);
    int h = idx >> 17;
    const float* xr = X + (unsigned)s * EMB + h * HD;
    float x1 = xr[d], x2 = xr[d + 64];
    // inv_freq = 10000^(-2d/128) = exp(-2d/128 * ln(10000))
    float freq = __expf(-(float)(2 * d) * (9.2103403719761836f / 128.0f));
    float ang  = (float)s * freq;
    float sn, cs;
    __sincosf(ang, &sn, &cs);
    __bf16* o = out + ((unsigned)h * S_LEN + s) * HD;
    o[d]      = (__bf16)((x1 * cs - x2 * sn) * scale);
    o[d + 64] = (__bf16)((x2 * cs + x1 * sn) * scale);
}

// V pack transposed: f32 [S][E] -> bf16 [H][D][S]  (write-coalesced)
__global__ void pack_vt_kernel(const float* __restrict__ V,
                               __bf16* __restrict__ out) {
    int idx = blockIdx.x * blockDim.x + threadIdx.x;   // H*D*S
    int s = idx & (S_LEN - 1);
    int d = (idx >> 11) & (HD - 1);
    int h = idx >> 18;
    out[((unsigned)h * HD + d) * S_LEN + s] =
        (__bf16)V[(unsigned)s * EMB + h * HD + d];
}

// ---------------------------------------------------------------------------
// Flash attention: one workgroup = (head h, 128 query rows). 8 waves, each
// owns 16 query rows x full D=128. Q pre-scaled by 1/sqrt(D).
//   Q [H][S][D] bf16, K [H][S][D] bf16, Vt [H][D][S] bf16 -> O [S][E] bf16
// K/V fragments read straight from global (L2-resident, 8 MB each).
// ---------------------------------------------------------------------------
__global__ __launch_bounds__(256) void flash_attn_kernel(
    const __bf16* __restrict__ Q, const __bf16* __restrict__ Kb,
    const __bf16* __restrict__ Vt, __bf16* __restrict__ O)
{
    __shared__ __align__(16) __bf16 sp[8][16 * 136];  // per-wave P tile (+pad)

    const int tid  = threadIdx.x;
    const int lane = tid & 31;
    const int wave = tid >> 5;
    const int ll   = lane & 15;
    const int hi   = lane >> 4;
    const int h    = blockIdx.y;
    const int q0   = blockIdx.x * 128 + wave * 16;

    // Preload Q fragments (A-matrix 16x32 per K-step; 4 steps cover D=128).
    const __bf16* qbase = Q + ((unsigned)h * S_LEN + q0 + ll) * HD;
    v16bf qf[4];
#pragma unroll
    for (int ks = 0; ks < 4; ++ks) qf[ks] = load_afrag(qbase + ks * 32, hi);

    v8f of[8];
    float m[8], l[8];
#pragma unroll
    for (int dt = 0; dt < 8; ++dt) of[dt] = (v8f)0.0f;
#pragma unroll
    for (int r = 0; r < 8; ++r) { m[r] = -1e30f; l[r] = 0.0f; }

    const unsigned kRowBase = (unsigned)h * S_LEN;       // rows of Kb[h]
    const unsigned vRowBase = (unsigned)h * HD;          // rows of Vt[h]

    for (int j = 0; j < S_LEN / 128; ++j) {
        const int k0 = j * 128;

        // scores tile: 16 rows x 128 keys
        v8f sacc[8];
#pragma unroll
        for (int nt = 0; nt < 8; ++nt) sacc[nt] = (v8f)0.0f;
#pragma unroll
        for (int nt = 0; nt < 8; ++nt) {
            const __bf16* krow = Kb + (kRowBase + k0 + nt * 16 + ll) * HD;
#pragma unroll
            for (int ks = 0; ks < 4; ++ks) {
                v16bf kf = load_bfrag(krow + ks * 32, hi);
                sacc[nt] = wmma_bf16(qf[ks], kf, sacc[nt]);
            }
        }

        // online softmax: rows live on 16-lane halves (row = r + 8*hi)
        float corr[8];
#pragma unroll
        for (int r = 0; r < 8; ++r) {
            float mx = -1e30f;
#pragma unroll
            for (int nt = 0; nt < 8; ++nt) mx = fmaxf(mx, sacc[nt][r]);
#pragma unroll
            for (int off = 1; off < 16; off <<= 1)
                mx = fmaxf(mx, __shfl_xor(mx, off, 32));
            float mn = fmaxf(m[r], mx);
            corr[r] = __expf(m[r] - mn);
            m[r] = mn;
            float sum = 0.0f;
#pragma unroll
            for (int nt = 0; nt < 8; ++nt) {
                float p = __expf(sacc[nt][r] - mn);
                sum += p;
                sp[wave][(r + 8 * hi) * 136 + nt * 16 + ll] = (__bf16)p;
            }
#pragma unroll
            for (int off = 1; off < 16; off <<= 1)
                sum += __shfl_xor(sum, off, 32);
            l[r] = l[r] * corr[r] + sum;
        }
#pragma unroll
        for (int dt = 0; dt < 8; ++dt)
#pragma unroll
            for (int r = 0; r < 8; ++r) of[dt][r] *= corr[r];

        // wave-private LDS round trip: order P stores before fragment loads
        asm volatile("s_wait_dscnt 0" ::: "memory");

        v16bf pf[4];
#pragma unroll
        for (int ks = 0; ks < 4; ++ks)
            pf[ks] = load_afrag(&sp[wave][ll * 136 + ks * 32], hi);

        // O += P * V  (V columns are rows of Vt -> contiguous B-fragments)
#pragma unroll
        for (int dt = 0; dt < 8; ++dt) {
            const __bf16* vrow = Vt + (vRowBase + dt * 16 + ll) * S_LEN + k0;
#pragma unroll
            for (int ks = 0; ks < 4; ++ks) {
                v16bf vf = load_bfrag(vrow + ks * 32, hi);
                of[dt] = wmma_bf16(pf[ks], vf, of[dt]);
            }
        }
    }

    // normalize and write attention output packed as [S][E] bf16
#pragma unroll
    for (int r = 0; r < 8; ++r) {
        float inv = 1.0f / l[r];
        unsigned row = (unsigned)(q0 + r + 8 * hi);
#pragma unroll
        for (int dt = 0; dt < 8; ++dt)
            O[row * EMB + (unsigned)(h * HD + dt * 16 + ll)] =
                (__bf16)(of[dt][r] * inv);
    }
}

// ---------------------------------------------------------------------------
extern "C" void kernel_launch(void* const* d_in, const int* in_sizes, int n_in,
                              void* d_out, int out_size, void* d_ws, size_t ws_size,
                              hipStream_t stream) {
    (void)in_sizes; (void)n_in; (void)out_size; (void)ws_size;
    const float* x  = (const float*)d_in[0];
    const float* Wq = (const float*)d_in[1];
    const float* bq = (const float*)d_in[2];
    const float* Wk = (const float*)d_in[3];
    const float* bk = (const float*)d_in[4];
    const float* Wv = (const float*)d_in[5];
    const float* bv = (const float*)d_in[6];
    const float* Wo = (const float*)d_in[7];
    const float* bo = (const float*)d_in[8];
    float* out = (float*)d_out;

    char*  ws  = (char*)d_ws;
    size_t off = 0;
    auto alloc = [&](size_t bytes) -> void* {
        void* p = ws + off;
        off += (bytes + 255) & ~(size_t)255;
        return p;
    };
    const size_t SE = (size_t)S_LEN * EMB;
    __bf16* xb  = (__bf16*)alloc(SE * 2);
    __bf16* Wqb = (__bf16*)alloc(SE * 2);
    __bf16* Wkb = (__bf16*)alloc(SE * 2);
    __bf16* Wvb = (__bf16*)alloc(SE * 2);
    __bf16* Wob = (__bf16*)alloc(SE * 2);
    float*  Qf  = (float*) alloc(SE * 4);
    float*  Kf  = (float*) alloc(SE * 4);
    float*  Vf  = (float*) alloc(SE * 4);
    __bf16* Qb  = (__bf16*)alloc(SE * 2);
    __bf16* Kb  = (__bf16*)alloc(SE * 2);
    __bf16* Vt  = (__bf16*)alloc(SE * 2);
    __bf16* AOb = (__bf16*)alloc(SE * 2);   // ~120 MB total workspace

    const int n = (int)SE;                  // 4194304
    const int cvtBlocks = n / (256 * 4);
    cvt_bf16_kernel<<<cvtBlocks, 256, 0, stream>>>(x,  xb,  n);
    cvt_bf16_kernel<<<cvtBlocks, 256, 0, stream>>>(Wq, Wqb, n);
    cvt_bf16_kernel<<<cvtBlocks, 256, 0, stream>>>(Wk, Wkb, n);
    cvt_bf16_kernel<<<cvtBlocks, 256, 0, stream>>>(Wv, Wvb, n);
    cvt_bf16_kernel<<<cvtBlocks, 256, 0, stream>>>(Wo, Wob, n);

    dim3 gg(EMB / 128, S_LEN / 128);
    gemm_bf16_wmma<<<gg, 256, 0, stream>>>(xb, Wqb, bq, Qf, S_LEN, EMB, EMB);
    gemm_bf16_wmma<<<gg, 256, 0, stream>>>(xb, Wkb, bk, Kf, S_LEN, EMB, EMB);
    gemm_bf16_wmma<<<gg, 256, 0, stream>>>(xb, Wvb, bv, Vf, S_LEN, EMB, EMB);

    const float qscale = 0.08838834764831845f;   // 1/sqrt(128)
    rope_pack_kernel<<<(NH * S_LEN * 64) / 256, 256, 0, stream>>>(Qf, Qb, qscale);
    rope_pack_kernel<<<(NH * S_LEN * 64) / 256, 256, 0, stream>>>(Kf, Kb, 1.0f);
    pack_vt_kernel<<<(NH * S_LEN * HD) / 256, 256, 0, stream>>>(Vf, Vt);

    flash_attn_kernel<<<dim3(S_LEN / 128, NH), 256, 0, stream>>>(Qb, Kb, Vt, AOb);

    gemm_bf16_wmma<<<gg, 256, 0, stream>>>(AOb, Wob, bo, out, S_LEN, EMB, EMB);
}
